// Correlation_28312424415883
// MI455X (gfx1250) — compile-verified, matrix operand order checked by reading, not run on hardware
//
#include <hip/hip_runtime.h>

#define BDIM 8
#define CDIM 128
#define HDIM 192
#define WDIM 192
#define MD   4
#define NDISP 81

typedef __attribute__((ext_vector_type(16))) __bf16 v16bf;
typedef __attribute__((ext_vector_type(8)))  __bf16 v8bf;
typedef __attribute__((ext_vector_type(8)))  float  v8f;

union Frag16 { v16bf v; v8bf h[2]; };

// round-to-nearest-even f32 -> bf16, packed pair into one dword
static __device__ __forceinline__ unsigned pack_bf16(float a, float b) {
    unsigned ua = __float_as_uint(a);
    unsigned ub = __float_as_uint(b);
    ua = (ua + 0x7FFFu + ((ua >> 16) & 1u)) >> 16;
    ub = (ub + 0x7FFFu + ((ub >> 16) & 1u)) >> 16;
    return (ua & 0xFFFFu) | (ub << 16);
}

__global__ __launch_bounds__(32)
void Correlation_28312424415883_kernel(const float* __restrict__ x1,
                                       const float* __restrict__ x2,
                                       float* __restrict__ out) {
    // LDS tiles, px-major, row stride 136 bf16 elems = 272B (16B multiple, kills
    // write-phase bank conflicts and keeps b128 fragment loads aligned)
    __shared__ __align__(16) unsigned short lA[16 * 136];  // x1 strip, 16px x 128c
    __shared__ __align__(16) unsigned short lB[32 * 136];  // x2 row,  32px x 128c
    __shared__ __align__(16) float          lP[16 * 34];   // P = A*B^T (16x32, padded)

    const int lane = threadIdx.x;      // 0..31
    const int r    = lane & 15;        // row / column within 16
    const int h    = lane >> 4;        // K-half select per WMMA 16-bit operand layout

    int blk = blockIdx.x;
    const int xs = blk % (WDIM / 16);  blk /= (WDIM / 16);
    const int y  = blk % HDIM;
    const int b  = blk / HDIM;
    const int x0 = xs * 16;

    const size_t planeStride = (size_t)HDIM * WDIM;
    const float* x1b = x1 + (size_t)b * CDIM * planeStride;
    const float* x2b = x2 + (size_t)b * CDIM * planeStride;

    // ---- stage A: x1[b, :, y, x0..x0+15] -> lA[px][c] as bf16 ----
    {
        const size_t rowOff = (size_t)y * WDIM + (x0 + r);
        for (int it = 0; it < 32; ++it) {
            const int c0 = it * 4 + 2 * h;  // lanes 0-15: c0=4it, lanes 16-31: 4it+2
            const float v0 = x1b[(size_t)c0 * planeStride + rowOff];
            const float v1 = x1b[(size_t)(c0 + 1) * planeStride + rowOff];
            *(unsigned*)&lA[r * 136 + c0] = pack_bf16(v0, v1);
        }
    }
    __syncthreads();

    // ---- preload all 4 A fragments (K-steps of 32) into VGPRs ----
    // 16-bit A operand: lane r   holds M=r, K = {k0..k0+7, k0+16..k0+23}
    //                   lane r+16 holds M=r, K = {k0+8..k0+15, k0+24..k0+31}
    Frag16 afrag[4];
#pragma unroll
    for (int s = 0; s < 4; ++s) {
        const unsigned short* p0 = &lA[r * 136 + s * 32 + 8 * h];
        afrag[s].h[0] = *(const v8bf*)p0;         // aligned ds_load_b128
        afrag[s].h[1] = *(const v8bf*)(p0 + 16);
    }

    const float inv_c = 1.0f / (float)CDIM;

    for (int dy = 0; dy < 9; ++dy) {
        const int yy = y + dy - MD;
        const int xx = x0 - MD + lane;
        const bool valid = (yy >= 0) && (yy < HDIM) && (xx >= 0) && (xx < WDIM);
        const size_t rowOff = (size_t)yy * WDIM + xx;   // used only when valid

        // prefetch next displacement row (c=0 plane) while we still have work
        const int yn = yy + 1;
        if (dy < 8 && yn >= 0 && yn < HDIM && xx >= 0 && xx < WDIM)
            __builtin_prefetch(x2b + (size_t)yn * WDIM + xx, 0, 0);

        // ---- stage B: x2p[b, :, yy, x0-4 .. x0+27] -> lB[px][c], zero-padded ----
        for (int it = 0; it < 64; ++it) {
            const int c0 = it * 2;
            float v0 = 0.0f, v1 = 0.0f;
            if (valid) {
                const float* p = x2b + (size_t)c0 * planeStride + rowOff;
                v0 = p[0];
                v1 = p[planeStride];
            }
            *(unsigned*)&lB[lane * 136 + c0] = pack_bf16(v0, v1);
        }
        __syncthreads();

        // ---- P = A (16x128) x B^T (128x32) via 2 col-tiles x 4 K-steps ----
        v8f acc0 = {0.f,0.f,0.f,0.f,0.f,0.f,0.f,0.f};
        v8f acc1 = {0.f,0.f,0.f,0.f,0.f,0.f,0.f,0.f};
#pragma unroll
        for (int s = 0; s < 4; ++s) {
            Frag16 bf0, bf1;
            const unsigned short* q0 = &lB[r * 136        + s * 32 + 8 * h];
            const unsigned short* q1 = &lB[(16 + r) * 136 + s * 32 + 8 * h];
            bf0.h[0] = *(const v8bf*)q0;  bf0.h[1] = *(const v8bf*)(q0 + 16);
            bf1.h[0] = *(const v8bf*)q1;  bf1.h[1] = *(const v8bf*)(q1 + 16);
            acc0 = __builtin_amdgcn_wmma_f32_16x16x32_bf16(
                       false, afrag[s].v, false, bf0.v, (short)0, acc0, false, false);
            acc1 = __builtin_amdgcn_wmma_f32_16x16x32_bf16(
                       false, afrag[s].v, false, bf1.v, (short)0, acc1, false, false);
        }

        // ---- spill P to LDS (D layout: vgpr k, lane -> M = k + 8*h, N = r) ----
        const int Mb = h * 8;
#pragma unroll
        for (int k = 0; k < 8; ++k) {
            lP[(Mb + k) * 34 + r]      = acc0[k];
            lP[(Mb + k) * 34 + 16 + r] = acc1[k];
        }
        __syncthreads();

        // ---- band extract: out[b, dy*9+dx, y, x0+i] = P[i][i+dx] / C ----
        {
            float* obase = out + (((size_t)b * NDISP + dy * 9) * HDIM + y) * WDIM + x0;
            for (int dx = h; dx < 9; dx += 2) {           // lanes split odd/even dx
                const float v = lP[r * 34 + r + dx] * inv_c;
                obase[(size_t)dx * planeStride + r] = v;  // 64B-coalesced per dx
            }
        }
        // next iteration's lB fill is fenced by the __syncthreads after it,
        // which also orders these lP reads before the next lP overwrite
        __syncthreads();
    }
}

extern "C" void kernel_launch(void* const* d_in, const int* in_sizes, int n_in,
                              void* d_out, int out_size, void* d_ws, size_t ws_size,
                              hipStream_t stream) {
    (void)in_sizes; (void)n_in; (void)out_size; (void)d_ws; (void)ws_size;
    const float* x1 = (const float*)d_in[0];
    const float* x2 = (const float*)d_in[1];
    float* out = (float*)d_out;
    const int nBlocks = (WDIM / 16) * HDIM * BDIM;   // 12 * 192 * 8 = 18432
    Correlation_28312424415883_kernel<<<nBlocks, 32, 0, stream>>>(x1, x2, out);
}
